// Glm4MoeMLPStack_40123584479298
// MI455X (gfx1250) — compile-verified, matrix operand order checked by reading, not run on hardware
//
#include <hip/hip_runtime.h>
#include <hip/hip_bf16.h>
#include <stdint.h>

// MoE MLP (SwiGLU) grouped GEMM for gfx1250 using v_wmma_f32_16x16x32_bf16.
// E=32 experts, H=2048, I=1408, T=32768 tokens (1024 per expert, contiguous).
// Software-pipelined: global loads for K-step k+1 issue into VGPRs while the
// WMMAs for K-step k execute out of LDS. Staging converts f32->bf16 in packed
// pairs (v_cvt_pk_bf16_f32 + ds_store_b32).

typedef __bf16 bf16;
typedef __attribute__((ext_vector_type(16))) __bf16 v16bf;
typedef __attribute__((ext_vector_type(8)))  __bf16 bf16x8;
typedef __attribute__((ext_vector_type(2)))  __bf16 bf16x2;
typedef __attribute__((ext_vector_type(8)))  float  v8f;

#define NE 32
#define NH 2048
#define NI 1408
#define NT 32768

__device__ __forceinline__ bf16 f2bf(float x) { return (bf16)x; }

__device__ __forceinline__ bf16x2 pack2(float a, float b) {
  bf16x2 p;
  p[0] = (bf16)a;
  p[1] = (bf16)b;
  return p;
}

// silu with v_rcp_f32 instead of IEEE divide (epilogue only; bf16 output).
__device__ __forceinline__ float fast_silu(float g) {
  float t = __expf(-g);
  return g * __builtin_amdgcn_rcpf(1.f + t);
}

// Build a 16-bf16 WMMA operand fragment from an LDS row holding 32 packed
// K-values. CDNA5 16-bit A/B layout: lanes 0-15 take K {0..7,16..23},
// lanes 16-31 take K {8..15,24..31}; kb = (lane>=16) ? 8 : 0.
__device__ __forceinline__ v16bf load_frag(const bf16* __restrict__ row, int kb) {
  bf16x8 lo = *(const bf16x8*)(row + kb);
  bf16x8 hi = *(const bf16x8*)(row + kb + 16);
  v16bf r;
#pragma unroll
  for (int i = 0; i < 8; ++i) { r[i] = lo[i]; r[i + 8] = hi[i]; }
  return r;
}

// Expert index of token m0 via prefix scan of group_sizes.
__device__ __forceinline__ int expert_of(const int* __restrict__ gs, int m0) {
  int e = 0, acc = 0;
#pragma unroll
  for (int i = 0; i < NE; ++i) {
    int nxt = acc + gs[i];
    if (m0 >= nxt) { acc = nxt; e = i + 1; }
  }
  return (e < NE) ? e : (NE - 1);
}

// ---------------------------------------------------------------------------
// Kernel 1: gate/up GEMMs + SwiGLU fused.  hidden[T,I](bf16) = silu(x@Wg)*(x@Wu)
// Block: 128 tokens x 128 cols, 256 threads = 8 waves (2x4 grid); each wave
// owns 64x32 -> 4x2 WMMA tiles per matrix (16 WMMA / K-step / wave).
// ---------------------------------------------------------------------------
__global__ __launch_bounds__(256) void moe_gateup_kernel(
    const float* __restrict__ x, const int* __restrict__ gs,
    const float* __restrict__ wg, const float* __restrict__ wu,
    bf16* __restrict__ hid) {
  __shared__ bf16 Xs[128][40];   // [m][k], pad 40 keeps ds_load_b128 16B aligned
  __shared__ bf16 Gs[128][40];   // [n][k] (transposed weight tile)
  __shared__ bf16 Us[128][40];

  const int tid = threadIdx.x;
  const int m0  = blockIdx.x * 128;
  const int n0  = blockIdx.y * 128;
  const int e   = expert_of(gs, m0);
  const size_t wb = (size_t)e * NH * NI;

  const int lane = tid & 31;
  const int wave = tid >> 5;
  const int wm   = (wave & 1) * 64;
  const int wn   = (wave >> 1) * 32;
  const int l16  = lane & 15;
  const int half = lane >> 4;
  const int kb   = half * 8;

  v8f ag[4][2], au[4][2];
#pragma unroll
  for (int i = 0; i < 4; ++i)
#pragma unroll
    for (int j = 0; j < 2; ++j) { ag[i][j] = (v8f){}; au[i][j] = (v8f){}; }

  // Register staging for the software pipeline (K-pair granularity).
  float2 rx[8];                  // X tile: 128x32 f32 = 2048 pairs / 256 thr
  float  rg[16], ru[16];         // W tiles: 32x128 f32 = 2048 pairs each

  auto fetch = [&](int k0) {
#pragma unroll
    for (int j = 0; j < 8; ++j) {            // X: m = p>>4, kpair = p&15; b64 loads
      int p = tid + j * 256;
      int m = p >> 4, kp = p & 15;
      rx[j] = *(const float2*)(x + (size_t)(m0 + m) * NH + k0 + kp * 2);
    }
#pragma unroll
    for (int j = 0; j < 8; ++j) {            // W: n = p&127, kpair = p>>7; n-coalesced
      int p = tid + j * 256;
      int n = p & 127, kp = p >> 7;
      size_t off = wb + (size_t)(k0 + kp * 2) * NI + n0 + n;
      rg[2 * j]     = wg[off];
      rg[2 * j + 1] = wg[off + NI];
      ru[2 * j]     = wu[off];
      ru[2 * j + 1] = wu[off + NI];
    }
  };

  auto stage = [&]() {                        // packed bf16 pair -> ds_store_b32
#pragma unroll
    for (int j = 0; j < 8; ++j) {
      int p = tid + j * 256;
      int m = p >> 4, kp = p & 15;
      *(bf16x2*)&Xs[m][kp * 2] = pack2(rx[j].x, rx[j].y);
    }
#pragma unroll
    for (int j = 0; j < 8; ++j) {
      int p = tid + j * 256;
      int n = p & 127, kp = p >> 7;
      *(bf16x2*)&Gs[n][kp * 2] = pack2(rg[2 * j], rg[2 * j + 1]);
      *(bf16x2*)&Us[n][kp * 2] = pack2(ru[2 * j], ru[2 * j + 1]);
    }
  };

  fetch(0);
  for (int k0 = 0; k0 < NH; k0 += 32) {
    stage();
    __syncthreads();
    if (k0 + 32 < NH) fetch(k0 + 32);        // overlap HBM latency with WMMAs

    v16bf a[4], bg[2], bu[2];
#pragma unroll
    for (int mt = 0; mt < 4; ++mt)
      a[mt] = load_frag(&Xs[wm + mt * 16 + l16][0], kb);
#pragma unroll
    for (int nt = 0; nt < 2; ++nt) {
      bg[nt] = load_frag(&Gs[wn + nt * 16 + l16][0], kb);
      bu[nt] = load_frag(&Us[wn + nt * 16 + l16][0], kb);
    }
#pragma unroll
    for (int mt = 0; mt < 4; ++mt)
#pragma unroll
      for (int nt = 0; nt < 2; ++nt) {
        ag[mt][nt] = __builtin_amdgcn_wmma_f32_16x16x32_bf16(
            false, a[mt], false, bg[nt], (short)0, ag[mt][nt], false, false);
        au[mt][nt] = __builtin_amdgcn_wmma_f32_16x16x32_bf16(
            false, a[mt], false, bu[nt], (short)0, au[mt][nt], false, false);
      }
    __syncthreads();                          // LDS consumed; safe to re-stage
  }

  // Epilogue: hidden = silu(g)*u, bf16 store.
  // C/D layout: vgpr r -> M = r + 8*half, lane%16 -> N.
#pragma unroll
  for (int mt = 0; mt < 4; ++mt)
#pragma unroll
    for (int nt = 0; nt < 2; ++nt)
#pragma unroll
      for (int r = 0; r < 8; ++r) {
        float g = ag[mt][nt][r];
        float u = au[mt][nt][r];
        int row = m0 + wm + mt * 16 + r + half * 8;
        int col = n0 + wn + nt * 16 + l16;
        hid[(size_t)row * NI + col] = f2bf(fast_silu(g) * u);
      }
}

// ---------------------------------------------------------------------------
// Kernel 2: down projection. out[T,H](f32) = hidden(bf16) @ Wd[e]
// ---------------------------------------------------------------------------
__global__ __launch_bounds__(256) void moe_down_kernel(
    const bf16* __restrict__ hid, const int* __restrict__ gs,
    const float* __restrict__ wd, float* __restrict__ out) {
  __shared__ bf16 Hs[128][40];   // [m][k]
  __shared__ bf16 Ds[128][40];   // [n][k]

  const int tid = threadIdx.x;
  const int m0  = blockIdx.x * 128;
  const int n0  = blockIdx.y * 128;
  const int e   = expert_of(gs, m0);
  const size_t wb = (size_t)e * NI * NH;

  const int lane = tid & 31;
  const int wave = tid >> 5;
  const int wm   = (wave & 1) * 64;
  const int wn   = (wave >> 1) * 32;
  const int l16  = lane & 15;
  const int half = lane >> 4;
  const int kb   = half * 8;

  v8f acc[4][2];
#pragma unroll
  for (int i = 0; i < 4; ++i)
#pragma unroll
    for (int j = 0; j < 2; ++j) acc[i][j] = (v8f){};

  uint32_t rh[8];                // hidden tile: bf16 pairs as dwords
  float    rd[16];               // Wd K-pairs

  auto fetch = [&](int k0) {
#pragma unroll
    for (int j = 0; j < 8; ++j) {            // 128x32 bf16 = 128x16 dwords
      int p = tid + j * 256;
      int m = p >> 4, kd = p & 15;
      rh[j] = *(const uint32_t*)(hid + (size_t)(m0 + m) * NI + k0 + kd * 2);
    }
#pragma unroll
    for (int j = 0; j < 8; ++j) {            // Wd: n = p&127, kpair = p>>7
      int p = tid + j * 256;
      int n = p & 127, kp = p >> 7;
      size_t off = wb + (size_t)(k0 + kp * 2) * NH + n0 + n;
      rd[2 * j]     = wd[off];
      rd[2 * j + 1] = wd[off + NH];
    }
  };

  auto stage = [&]() {
#pragma unroll
    for (int j = 0; j < 8; ++j) {
      int p = tid + j * 256;
      *(uint32_t*)&Hs[p >> 4][(p & 15) * 2] = rh[j];
    }
#pragma unroll
    for (int j = 0; j < 8; ++j) {
      int p = tid + j * 256;
      int n = p & 127, kp = p >> 7;
      *(bf16x2*)&Ds[n][kp * 2] = pack2(rd[2 * j], rd[2 * j + 1]);
    }
  };

  fetch(0);
  for (int k0 = 0; k0 < NI; k0 += 32) {
    stage();
    __syncthreads();
    if (k0 + 32 < NI) fetch(k0 + 32);

    v16bf a[4], b[2];
#pragma unroll
    for (int mt = 0; mt < 4; ++mt)
      a[mt] = load_frag(&Hs[wm + mt * 16 + l16][0], kb);
#pragma unroll
    for (int nt = 0; nt < 2; ++nt)
      b[nt] = load_frag(&Ds[wn + nt * 16 + l16][0], kb);
#pragma unroll
    for (int mt = 0; mt < 4; ++mt)
#pragma unroll
      for (int nt = 0; nt < 2; ++nt)
        acc[mt][nt] = __builtin_amdgcn_wmma_f32_16x16x32_bf16(
            false, a[mt], false, b[nt], (short)0, acc[mt][nt], false, false);
    __syncthreads();
  }

#pragma unroll
  for (int mt = 0; mt < 4; ++mt)
#pragma unroll
    for (int nt = 0; nt < 2; ++nt)
#pragma unroll
      for (int r = 0; r < 8; ++r) {
        int row = m0 + wm + mt * 16 + r + half * 8;
        int col = n0 + wn + nt * 16 + l16;
        out[(size_t)row * NH + col] = acc[mt][nt][r];
      }
}

extern "C" void kernel_launch(void* const* d_in, const int* in_sizes, int n_in,
                              void* d_out, int out_size, void* d_ws, size_t ws_size,
                              hipStream_t stream) {
  const float* x  = (const float*)d_in[0];
  const int*   gs = (const int*)d_in[1];
  const float* wg = (const float*)d_in[2];
  const float* wu = (const float*)d_in[3];
  const float* wd = (const float*)d_in[4];
  float* out = (float*)d_out;
  bf16*  hid = (bf16*)d_ws;  // T*I bf16 = ~92 MB scratch

  dim3 blk(256);
  dim3 gA(NT / 128, NI / 128);  // 256 x 11
  moe_gateup_kernel<<<gA, blk, 0, stream>>>(x, gs, wg, wu, hid);
  dim3 gB(NT / 128, NH / 128);  // 256 x 16
  moe_down_kernel<<<gB, blk, 0, stream>>>(hid, gs, wd, out);
}
